// GNN_9466107921144
// MI455X (gfx1250) — compile-verified, matrix operand order checked by reading, not run on hardware
//
#include <hip/hip_runtime.h>

#define BATCH 2048
#define NNODE 64
#define CDIM  512
#define KNB   4
#define BN_EPS 1e-5f

typedef __attribute__((ext_vector_type(16))) __bf16 v16bf;
typedef __attribute__((ext_vector_type(8)))  float  v8f;

union FragBF { v16bf v; unsigned int u[8]; };

__device__ __forceinline__ unsigned short f2bf(float f) {
    unsigned int u = __float_as_uint(f);
    u += 0x7FFFu + ((u >> 16) & 1u);           // round-to-nearest-even
    return (unsigned short)(u >> 16);
}

// Low 32 bits of a generic pointer to LDS == wave-relative LDS byte offset.
__device__ __forceinline__ unsigned lds_off(const void* p) {
    return (unsigned)(unsigned long long)p;
}

// Async-copy one 512x32 bf16 weight K-panel (32 KB) global->LDS.
// 2048 x 16B transfers, 4 per thread (512 threads). ASYNCcnt-tracked.
__device__ __forceinline__ void issue_panel(const unsigned short* __restrict__ wsrc,
                                            unsigned lds_base, int kk, int tid) {
#pragma unroll
    for (int c = 0; c < 4; ++c) {
        int j = tid * 4 + c;                  // 0..2047
        int n = j >> 2, q = j & 3;
        const void* g = (const void*)(wsrc + (size_t)n * CDIM + kk * 32 + q * 8);
        unsigned l = lds_base + (unsigned)(n * 64 + q * 16);
        asm volatile("global_load_async_to_lds_b128 %0, %1, off"
                     :: "v"(l), "v"(g) : "memory");
    }
}

__device__ __forceinline__ void wait_async0() {
    asm volatile("s_wait_asynccnt 0" ::: "memory");
}

// Load a 16x32 bf16 WMMA operand fragment from an LDS row-major bf16 array.
__device__ __forceinline__ v16bf ld_frag(const unsigned short* lds, int row,
                                         int kbase, int stride, int half) {
    FragBF f;
#pragma unroll
    for (int j = 0; j < 8; ++j) {
        int k0 = 2 * j + ((j >= 4) ? 8 : 0) + half * 8;
        f.u[j] = *(const unsigned int*)(lds + row * stride + kbase + k0);
    }
    return f.v;
}

__global__ void gnn_zero_kernel(float* acc) {
    acc[threadIdx.x] = 0.0f;   // 128 threads: gsum[64] + gsq[64]
}

// One-time f32 -> bf16 conversion of both weight matrices into workspace.
__global__ __launch_bounds__(256) void gnn_cvtw_kernel(
    const float* __restrict__ Wv, const float* __restrict__ Wu,
    unsigned short* __restrict__ wbf)
{
    size_t i = (size_t)blockIdx.x * 256 + threadIdx.x;      // 0 .. 2*512*512-1
    const size_t half = (size_t)CDIM * CDIM;
    float v = (i < half) ? Wv[i] : Wu[i - half];
    wbf[i] = f2bf(v);
}

__global__ __launch_bounds__(512) void gnn_main_kernel(
    const float* __restrict__ x,  const unsigned short* __restrict__ wbf,
    const float* __restrict__ bu, const float* __restrict__ bv,
    float* __restrict__ z_out, float* __restrict__ gsum, float* __restrict__ gsq)
{
    __shared__ unsigned short xbf[NNODE * CDIM];      // 64 KB  x_b as bf16
    __shared__ unsigned short hvT[CDIM * NNODE];      // 64 KB  hv transposed [c][node]
    __shared__ unsigned short wpanel[2][CDIM * 32];   // 2x32 KB double-buffered weight panels
    __shared__ unsigned short conbf[NNODE * NNODE];   //  8 KB  normalized adjacency bf16
    __shared__ float thrv[NNODE];
    __shared__ float dis[NNODE];
    __shared__ float psum[NNODE];
    __shared__ float psq[NNODE];

    float* sim = (float*)&wpanel[0][0];               // 16 KB, dead before first panel copy

    const int b    = blockIdx.x;
    const int tid  = threadIdx.x;
    const int wv   = tid >> 5;        // wave 0..15
    const int lane = tid & 31;
    const int half = lane >> 4;
    const int l16  = lane & 15;

    const unsigned short* wv_bf = wbf;                        // Wv in bf16
    const unsigned short* wu_bf = wbf + (size_t)CDIM * CDIM;  // Wu in bf16
    const unsigned poff0 = lds_off(&wpanel[0][0]);
    const unsigned poff1 = lds_off(&wpanel[1][0]);

    // ---- 1) stage x_b into LDS as bf16 ----
    const float* xb = x + (size_t)b * NNODE * CDIM;
    for (int i = tid; i < NNODE * CDIM; i += 512) xbf[i] = f2bf(xb[i]);
    __syncthreads();

    // ---- 2) sim = x_b x_bT via WMMA: wave w owns tile (ti=w&3, tj=w>>2) ----
    {
        const int ti = wv & 3, tj = wv >> 2;
        v8f acc = {};
        for (int kk = 0; kk < CDIM / 32; ++kk) {
            v16bf a  = ld_frag(xbf, ti * 16 + l16, kk * 32, CDIM, half);
            v16bf bm = ld_frag(xbf, tj * 16 + l16, kk * 32, CDIM, half);
            acc = __builtin_amdgcn_wmma_f32_16x16x32_bf16(false, a, false, bm,
                                                          (short)0, acc, false, false);
        }
#pragma unroll
        for (int r = 0; r < 8; ++r) {
            int m = ti * 16 + half * 8 + r;
            int n = tj * 16 + l16;
            sim[m * NNODE + n] = acc[r];
        }
    }
    __syncthreads();

    // ---- 3) per-row top-4 threshold, degree, D^{-1/2} ----
    if (tid < NNODE) {
        unsigned long long used = 0ull;
        float thr = -3.0e38f;
        for (int it = 0; it < KNB; ++it) {
            float best = -3.0e38f; int bj = 0;
            for (int j = 0; j < NNODE; ++j) {
                float v = sim[tid * NNODE + j];
                if (!((used >> j) & 1ull) && v > best) { best = v; bj = j; }
            }
            used |= 1ull << bj;
            thr = best;
        }
        int deg = 0;
        for (int j = 0; j < NNODE; ++j) deg += (sim[tid * NNODE + j] >= thr) ? 1 : 0;
        thrv[tid] = thr;
        dis[tid]  = rsqrtf((float)deg);
        psum[tid] = 0.0f; psq[tid] = 0.0f;
    }
    __syncthreads();

    // ---- 4) con = adj * dis_i * dis_j, stored bf16 ----
    for (int idx = tid; idx < NNODE * NNODE; idx += 512) {
        int i = idx >> 6, j = idx & 63;
        float v = (sim[idx] >= thrv[i]) ? dis[i] * dis[j] : 0.0f;
        conbf[idx] = f2bf(v);
    }
    __syncthreads();                                  // sim dead; panels may now use buffer 0

    // per-wave output tiling for the 64x512 GEMMs: 4 m-tiles x 32 n-tiles
    const int tm  = wv & 3;
    const int tn0 = (wv >> 2) * 8;

    // ---- 5) hv = x_b @ WvT + bv, double-buffered async weight panels ----
    v8f hacc[8] = {};
    issue_panel(wv_bf, poff0, 0, tid);
    for (int kk = 0; kk < CDIM / 32; ++kk) {
        wait_async0();                                // my copies of current panel done
        __syncthreads();                              // everyone's copies done / prev readers done
        if (kk + 1 < CDIM / 32)
            issue_panel(wv_bf, (kk & 1) ? poff0 : poff1, kk + 1, tid);
        const unsigned short* pan = wpanel[kk & 1];
        v16bf a = ld_frag(xbf, tm * 16 + l16, kk * 32, CDIM, half);
#pragma unroll
        for (int t = 0; t < 8; ++t) {
            int n = (tn0 + t) * 16 + l16;
            v16bf bm = ld_frag(pan, n, 0, 32, half);
            hacc[t] = __builtin_amdgcn_wmma_f32_16x16x32_bf16(false, a, false, bm,
                                                              (short)0, hacc[t], false, false);
        }
    }
    // add bv, store transposed bf16 hv into LDS: hvT[c][node]
#pragma unroll
    for (int t = 0; t < 8; ++t) {
        int n = (tn0 + t) * 16 + l16;
        float bvn = bv[n];
#pragma unroll
        for (int r = 0; r < 8; ++r) {
            int m = tm * 16 + half * 8 + r;
            hvT[n * NNODE + m] = f2bf(hacc[t][r] + bvn);
        }
    }
    __syncthreads();

    // ---- 6) z = con @ hv + x_b @ WuT + bu ----
    v8f zacc[8] = {};
    issue_panel(wu_bf, poff0, 0, tid);                // overlap with agg GEMM below
#pragma unroll
    for (int kk = 0; kk < 2; ++kk) {                  // agg: K = 64 nodes
        v16bf a = ld_frag(conbf, tm * 16 + l16, kk * 32, NNODE, half);
#pragma unroll
        for (int t = 0; t < 8; ++t) {
            int n = (tn0 + t) * 16 + l16;
            v16bf bm = ld_frag(hvT, n, kk * 32, NNODE, half);
            zacc[t] = __builtin_amdgcn_wmma_f32_16x16x32_bf16(false, a, false, bm,
                                                              (short)0, zacc[t], false, false);
        }
    }
    for (int kk = 0; kk < CDIM / 32; ++kk) {          // + x @ WuT
        wait_async0();
        __syncthreads();
        if (kk + 1 < CDIM / 32)
            issue_panel(wu_bf, (kk & 1) ? poff0 : poff1, kk + 1, tid);
        const unsigned short* pan = wpanel[kk & 1];
        v16bf a = ld_frag(xbf, tm * 16 + l16, kk * 32, CDIM, half);
#pragma unroll
        for (int t = 0; t < 8; ++t) {
            int n = (tn0 + t) * 16 + l16;
            v16bf bm = ld_frag(pan, n, 0, 32, half);
            zacc[t] = __builtin_amdgcn_wmma_f32_16x16x32_bf16(false, a, false, bm,
                                                              (short)0, zacc[t], false, false);
        }
    }

    // add bu; accumulate per-node BN partial sums; spill z
    float* zb = z_out + (size_t)b * NNODE * CDIM;
#pragma unroll
    for (int t = 0; t < 8; ++t) {
        int n = (tn0 + t) * 16 + l16;
        float bun = bu[n];
#pragma unroll
        for (int r = 0; r < 8; ++r) zacc[t][r] += bun;
    }
#pragma unroll
    for (int r = 0; r < 8; ++r) {
        int m = tm * 16 + half * 8 + r;
        float s = 0.0f, q = 0.0f;
#pragma unroll
        for (int t = 0; t < 8; ++t) { float v = zacc[t][r]; s += v; q += v * v; }
        atomicAdd(&psum[m], s);     // ds_add_f32
        atomicAdd(&psq[m], q);
    }
#pragma unroll
    for (int t = 0; t < 8; ++t) {
        int n = (tn0 + t) * 16 + l16;
#pragma unroll
        for (int r = 0; r < 8; ++r) {
            int m = tm * 16 + half * 8 + r;
            zb[m * CDIM + n] = zacc[t][r];
        }
    }
    __syncthreads();
    if (tid < NNODE) {
        atomicAdd(&gsum[tid], psum[tid]);
        atomicAdd(&gsq[tid],  psq[tid]);
    }
}

__global__ void gnn_stats_kernel(const float* __restrict__ gsum, const float* __restrict__ gsq,
                                 const float* __restrict__ bn_w, const float* __restrict__ bn_b,
                                 float* __restrict__ params)   // [0:64)=scale, [64:128)=shift
{
    int m = threadIdx.x;
    const float cnt = (float)BATCH * (float)CDIM;
    float mean = gsum[m] / cnt;
    float var  = gsq[m] / cnt - mean * mean;
    float a = bn_w[m] * rsqrtf(var + BN_EPS);
    params[m]         = a;
    params[m + NNODE] = bn_b[m] - mean * a;
}

__global__ __launch_bounds__(256) void gnn_finalize_kernel(
    const float* __restrict__ x, const float* __restrict__ z,
    const float* __restrict__ params, float* __restrict__ out)
{
    size_t i = (size_t)blockIdx.x * blockDim.x + threadIdx.x;   // float4 index
    int m = (int)((i >> 7) & (NNODE - 1));                      // (i*4/512)%64
    float a = params[m], sh = params[m + NNODE];
    float4 xv = ((const float4*)x)[i];
    float4 zv = ((const float4*)z)[i];
    float4 o;
    o.x = fmaxf(xv.x + zv.x * a + sh, 0.0f);
    o.y = fmaxf(xv.y + zv.y * a + sh, 0.0f);
    o.z = fmaxf(xv.z + zv.z * a + sh, 0.0f);
    o.w = fmaxf(xv.w + zv.w * a + sh, 0.0f);
    ((float4*)out)[i] = o;
}

extern "C" void kernel_launch(void* const* d_in, const int* in_sizes, int n_in,
                              void* d_out, int out_size, void* d_ws, size_t ws_size,
                              hipStream_t stream) {
    const float* x    = (const float*)d_in[0];
    const float* Wu   = (const float*)d_in[1];
    const float* bu   = (const float*)d_in[2];
    const float* Wv   = (const float*)d_in[3];
    const float* bv   = (const float*)d_in[4];
    const float* bn_w = (const float*)d_in[5];
    const float* bn_b = (const float*)d_in[6];
    float* out = (float*)d_out;

    const size_t z_elems = (size_t)BATCH * NNODE * CDIM;
    float* z      = (float*)d_ws;
    float* gsum   = z + z_elems;
    float* gsq    = gsum + NNODE;
    float* params = gsq + NNODE;
    unsigned short* wbf = (unsigned short*)(params + 2 * NNODE);  // 2*512*512 bf16 = 1 MB

    gnn_cvtw_kernel<<<(2 * CDIM * CDIM) / 256, 256, 0, stream>>>(Wv, Wu, wbf);
    gnn_zero_kernel<<<1, 2 * NNODE, 0, stream>>>(gsum);   // zeros gsum+gsq
    gnn_main_kernel<<<BATCH, 512, 0, stream>>>(x, wbf, bu, bv, z, gsum, gsq);
    gnn_stats_kernel<<<1, NNODE, 0, stream>>>(gsum, gsq, bn_w, bn_b, params);
    const size_t n4 = z_elems / 4;
    gnn_finalize_kernel<<<(unsigned)(n4 / 256), 256, 0, stream>>>(x, z, params, out);
}
